// MultiheadAttention_68762426409199
// MI455X (gfx1250) — matrix operand order repaired and checked by reading, not compiled, not run on hardware
//
#include <hip/hip_runtime.h>

// MI455X fused multihead-attention-over-encoder-steps.
// Roofline: ~515 MB mandatory reads (Wk/Wv dominate) -> 22 us @ 23.3 TB/s.
// 13.2 GFLOP fp32 => need >=600 TF/s matrix rate to stay memory bound ->
// bf16 WMMA with hi/lo split (bf16x3) for fp32-class accuracy.
// Grid: (N * 2) workgroups; each handles one node n and 32 of the 64 batch
// rows with 8 wave32s (2 row-tiles x 4 column-quarters of the F=256 dim).
// Weight chunks are software-pipelined: global loads for chunk c+1 are issued
// into registers before the WMMAs of chunk c, with double-buffered packed
// bf16 hi/lo weight tiles in LDS.

#define DEVINL __device__ __forceinline__

typedef __attribute__((ext_vector_type(16))) __bf16       v16bf;
typedef __attribute__((ext_vector_type(2)))  __bf16       v2bf;
typedef __attribute__((ext_vector_type(8)))  float        v8f;
typedef __attribute__((ext_vector_type(4)))  unsigned int u32x4;

union Frag {
  v16bf bf;
  u32x4 u[2];
};

constexpr int Bb = 64, Hh = 24, Nn = 32, Ff = 256;
constexpr int AP = 132;  // A LDS pitch (dwords): 128 packed pairs + 4 pad (16B aligned rows)
constexpr int BP = 36;   // B LDS pitch (dwords): 32 packed pairs + 4 pad

// Pack two fp32 -> packed bf16 (RNE), low half = first arg.
#if __has_builtin(__builtin_amdgcn_cvt_pk_bf16_f32)
DEVINL unsigned pk2(float a, float b) {
  v2bf r = __builtin_amdgcn_cvt_pk_bf16_f32(a, b);
  return __builtin_bit_cast(unsigned, r);
}
#else
DEVINL unsigned pk2(float a, float b) {
  __bf16 ah = (__bf16)a, bh = (__bf16)b;
  return ((unsigned)__builtin_bit_cast(unsigned short, bh) << 16) |
         (unsigned)__builtin_bit_cast(unsigned short, ah);
}
#endif

// Split fp32 pair (a,b) into packed bf16 hi parts and packed bf16 lo residuals.
DEVINL void split2(float a, float b, unsigned &hp, unsigned &lp) {
  hp = pk2(a, b);
  float ar = __builtin_bit_cast(float, hp << 16);          // hi(a) back to f32
  float br = __builtin_bit_cast(float, hp & 0xffff0000u);  // hi(b) back to f32
  lp = pk2(a - ar, b - br);                                // exact residuals
}

__global__ __launch_bounds__(256, 1)
void mha_fused(const float* __restrict__ Q,  const float* __restrict__ K,
               const float* __restrict__ V,  const float* __restrict__ Wq,
               const float* __restrict__ bq, const float* __restrict__ Wk,
               const float* __restrict__ bk, const float* __restrict__ Wv,
               const float* __restrict__ bv, float* __restrict__ heads,
               float* __restrict__ scores) {
  __shared__ __align__(16) unsigned Ahi[32 * AP], Alo[32 * AP];     // activations, split bf16
  __shared__ __align__(16) unsigned Bhi0[256 * BP], Blo0[256 * BP]; // weight chunk buf 0
  __shared__ __align__(16) unsigned Bhi1[256 * BP], Blo1[256 * BP]; // weight chunk buf 1
  __shared__ float part[16 * 8];     // cross-wave logit partials
  __shared__ float logitsS[32 * 24]; // logits -> scores, per local batch row

  const int tid  = threadIdx.x;
  const int lane = tid & 31, wid = tid >> 5;
  const int rt   = wid >> 2;        // row-tile (16 rows each)
  const int cq   = wid & 3;         // column quarter (64 cols each)
  const int g0   = cq * 64;
  const int lan  = lane & 15, hs = lane >> 4;
  const int nIdx = blockIdx.x >> 1;
  const int bG0  = (blockIdx.x & 1) * 32;

  // Load a 32x256 fp32 activation slice -> packed hi/lo bf16 pairs in LDS.
  auto loadA = [&](const float* src, long rowStride) {
    for (int u = tid; u < 2048; u += 256) {
      int b = u >> 6;
      int f = (u & 63) << 2;
      float4 x = *reinterpret_cast<const float4*>(src + (long)b * rowStride + f);
      unsigned h0, l0, h1, l1;
      split2(x.x, x.y, h0, l0);
      split2(x.z, x.w, h1, l1);
      int pi = b * AP + (f >> 1);
      Ahi[pi] = h0; Ahi[pi + 1] = h1;
      Alo[pi] = l0; Alo[pi + 1] = l1;
    }
  };

  // Issue global loads for one 64-row weight chunk into registers (8 units/thread,
  // each unit = two adjacent K rows so pairs can be packed at commit time).
  auto issueW = [&](const float* w, int f0, float4* ra, float4* rc) {
#pragma unroll
    for (int i = 0; i < 8; ++i) {
      int p = (tid >> 6) + i * 4;     // K-pair within chunk (0..31)
      int g = (tid & 63) << 2;        // output column quad (coalesced across lanes)
      const float* r = w + (long)(f0 + 2 * p) * Ff + g;
      ra[i] = *reinterpret_cast<const float4*>(r);
      rc[i] = *reinterpret_cast<const float4*>(r + Ff);
    }
  };

  // Convert staged registers -> packed bf16 hi/lo WMMA B-operand layout in LDS.
  auto commitW = [&](unsigned* BH, unsigned* BL, const float4* ra, const float4* rc) {
#pragma unroll
    for (int i = 0; i < 8; ++i) {
      int p = (tid >> 6) + i * 4;
      int g = (tid & 63) << 2;
      unsigned hp, lp;
      split2(ra[i].x, rc[i].x, hp, lp); BH[(g + 0) * BP + p] = hp; BL[(g + 0) * BP + p] = lp;
      split2(ra[i].y, rc[i].y, hp, lp); BH[(g + 1) * BP + p] = hp; BL[(g + 1) * BP + p] = lp;
      split2(ra[i].z, rc[i].z, hp, lp); BH[(g + 2) * BP + p] = hp; BL[(g + 2) * BP + p] = lp;
      split2(ra[i].w, rc[i].w, hp, lp); BH[(g + 3) * BP + p] = hp; BL[(g + 3) * BP + p] = lp;
    }
  };

  // Multiply one 64-wide K chunk into the wave's 16x64 accumulator tile set.
  auto mmaChunk = [&](int f0, v8f acc[4], const unsigned* BH, const unsigned* BL) {
#pragma unroll
    for (int s = 0; s < 2; ++s) {            // two K=32 WMMA steps per chunk
      Frag ah, al;
      int rowA = (rt * 16 + lan) * AP;
      int pA = (f0 >> 1) + s * 16 + hs * 4;  // A layout: lanes>=16 hold K+8..15 / K+24..31
      ah.u[0] = *reinterpret_cast<const u32x4*>(&Ahi[rowA + pA]);
      ah.u[1] = *reinterpret_cast<const u32x4*>(&Ahi[rowA + pA + 8]);
      al.u[0] = *reinterpret_cast<const u32x4*>(&Alo[rowA + pA]);
      al.u[1] = *reinterpret_cast<const u32x4*>(&Alo[rowA + pA + 8]);
      int pB = s * 16 + hs * 4;              // B layout: lanes>=16 hold K+8..15 / K+24..31
#pragma unroll
      for (int t = 0; t < 4; ++t) {
        int rowB = (g0 + t * 16 + lan) * BP;
        Frag bh_, bl_;
        bh_.u[0] = *reinterpret_cast<const u32x4*>(&BH[rowB + pB]);
        bh_.u[1] = *reinterpret_cast<const u32x4*>(&BH[rowB + pB + 8]);
        bl_.u[0] = *reinterpret_cast<const u32x4*>(&BL[rowB + pB]);
        bl_.u[1] = *reinterpret_cast<const u32x4*>(&BL[rowB + pB + 8]);
        // bf16x3: Ahi*Bhi + Ahi*Blo + Alo*Bhi  (fp32 accumulate)
        acc[t] = __builtin_amdgcn_wmma_f32_16x16x32_bf16(false, ah.bf, false, bh_.bf,
                                                         (short)0, acc[t], false, false);
        acc[t] = __builtin_amdgcn_wmma_f32_16x16x32_bf16(false, ah.bf, false, bl_.bf,
                                                         (short)0, acc[t], false, false);
        acc[t] = __builtin_amdgcn_wmma_f32_16x16x32_bf16(false, al.bf, false, bh_.bf,
                                                         (short)0, acc[t], false, false);
      }
    }
  };

  // Full 32x256 x 256x256 GEMM + column bias into acc, pipelined over 4 K-chunks.
  auto runGemm = [&](const float* X, long xs, const float* W, const float* bias, v8f acc[4]) {
    float4 ra[8], rc[8];
    __syncthreads();           // previous users of A/B buffers are done
    issueW(W, 0, ra, rc);      // chunk 0 loads in flight while A is prepared
    loadA(X, xs);
    for (int c = 0; c < 4; ++c) {
      unsigned* BH = (c & 1) ? Bhi1 : Bhi0;
      unsigned* BL = (c & 1) ? Blo1 : Blo0;
      __syncthreads();                       // buf[c&1] readers (compute c-2) done
      commitW(BH, BL, ra, rc);               // convert + store chunk c
      if (c < 3) issueW(W, (c + 1) * 64, ra, rc);  // next chunk hidden behind WMMAs
      if (c < 2 && tid < 64)                 // prefetch two chunks ahead
        __builtin_prefetch(W + (long)((c + 2) * 64 + tid) * Ff, 0, 3);
      __syncthreads();                       // commits visible to all waves
      mmaChunk(c * 64, acc, BH, BL);
    }
#pragma unroll
    for (int t = 0; t < 4; ++t) {
      float bb = bias[g0 + t * 16 + lan];
#pragma unroll
      for (int v = 0; v < 8; ++v) acc[t][v] += bb;
    }
  };

  v8f qw[4], acc[4];
  const v8f vz = {};

  // ---- QWq = Q[:,n,:] @ Wq[n] + bq[n], kept in registers ----
#pragma unroll
  for (int t = 0; t < 4; ++t) acc[t] = vz;
  runGemm(Q + ((long)bG0 * Nn + nIdx) * Ff, (long)Nn * Ff,
          Wq + (long)nIdx * Ff * Ff, bq + (long)nIdx * Ff, acc);
#pragma unroll
  for (int t = 0; t < 4; ++t) qw[t] = acc[t];

  // ---- pass 1: logits[b,h] = dot(QWq[b,:], K[:,h,n,:] @ Wk[n,h] + bk) ----
  for (int h = 0; h < Hh; ++h) {
#pragma unroll
    for (int t = 0; t < 4; ++t) acc[t] = vz;
    long whOff = (long)nIdx * Hh + h;
    runGemm(K + (((long)bG0 * Hh + h) * Nn + nIdx) * Ff, (long)Hh * Nn * Ff,
            Wk + whOff * Ff * Ff, bk + whOff * Ff, acc);
    float pd[8];
#pragma unroll
    for (int v = 0; v < 8; ++v) {
      float x = acc[0][v] * qw[0][v] + acc[1][v] * qw[1][v] +
                acc[2][v] * qw[2][v] + acc[3][v] * qw[3][v];
      x += __shfl_xor(x, 1, 32);   // reduce over the 16 columns held in this half-wave
      x += __shfl_xor(x, 2, 32);
      x += __shfl_xor(x, 4, 32);
      x += __shfl_xor(x, 8, 32);
      pd[v] = x;
    }
    if (lan == 0) {
#pragma unroll
      for (int v = 0; v < 8; ++v) part[(wid * 2 + hs) * 8 + v] = pd[v];
    }
    __syncthreads();
    if (tid < 32) {              // finish: sum the 4 column-quarter waves
      int b = tid, rtb = b >> 4, m = b & 15;
      float s = 0.f;
#pragma unroll
      for (int c = 0; c < 4; ++c)
        s += part[(((rtb * 4 + c) * 2) + (m >> 3)) * 8 + (m & 7)];
      logitsS[b * 24 + h] = s;
    }
  }

  // ---- softmax over H=24, write scores output ----
  __syncthreads();
  if (tid < 32) {
    int b = tid;
    float mx = -3.0e38f;
    for (int h = 0; h < Hh; ++h) mx = fmaxf(mx, logitsS[b * 24 + h]);
    float e[Hh], s = 0.f;
    for (int h = 0; h < Hh; ++h) { e[h] = __expf(logitsS[b * 24 + h] - mx); s += e[h]; }
    float inv = 1.0f / s;
    long bG = bG0 + b;
    for (int h = 0; h < Hh; ++h) {
      float sc = e[h] * inv;
      logitsS[b * 24 + h] = sc;
      scores[(bG * Nn + nIdx) * Hh + h] = sc;
    }
  }
  __syncthreads();

  // ---- pass 2: heads[b,:] = sum_h scores[b,h] * (V[:,h,n,:] @ Wv[n,h] + bv) ----
#pragma unroll
  for (int t = 0; t < 4; ++t) qw[t] = vz;   // reuse QWq registers as head accumulators
  for (int h = 0; h < Hh; ++h) {
#pragma unroll
    for (int t = 0; t < 4; ++t) acc[t] = vz;
    long whOff = (long)nIdx * Hh + h;
    runGemm(V + (((long)bG0 * Hh + h) * Nn + nIdx) * Ff, (long)Hh * Nn * Ff,
            Wv + whOff * Ff * Ff, bv + whOff * Ff, acc);
    int rowl = rt * 16 + hs * 8;
#pragma unroll
    for (int v = 0; v < 8; ++v) {
      float sc = logitsS[(rowl + v) * 24 + h];
#pragma unroll
      for (int t = 0; t < 4; ++t) qw[t][v] += sc * acc[t][v];
    }
  }

  // ---- store heads ----
#pragma unroll
  for (int t = 0; t < 4; ++t) {
#pragma unroll
    for (int v = 0; v < 8; ++v) {
      long bG = bG0 + rt * 16 + hs * 8 + v;
      heads[(bG * Nn + nIdx) * Ff + (g0 + t * 16 + lan)] = qw[t][v];
    }
  }
}

extern "C" void kernel_launch(void* const* d_in, const int* in_sizes, int n_in,
                              void* d_out, int out_size, void* d_ws, size_t ws_size,
                              hipStream_t stream) {
  (void)in_sizes; (void)n_in; (void)out_size; (void)d_ws; (void)ws_size;
  const float* Q  = (const float*)d_in[0];
  const float* K  = (const float*)d_in[1];
  const float* V  = (const float*)d_in[2];
  const float* Wq = (const float*)d_in[3];
  const float* bq = (const float*)d_in[4];
  const float* Wk = (const float*)d_in[5];
  const float* bk = (const float*)d_in[6];
  const float* Wv = (const float*)d_in[7];
  const float* bv = (const float*)d_in[8];
  float* heads   = (float*)d_out;
  float* scoresO = heads + (size_t)Bb * Nn * Ff;
  mha_fused<<<dim3(Nn * 2), dim3(256), 0, stream>>>(Q, K, V, Wq, bq, Wk, bk, Wv, bv,
                                                    heads, scoresO);
}
